// MoEKANConvBase_71983651881055
// MI455X (gfx1250) — matrix operand order repaired and checked by reading, not compiled, not run on hardware
//
#include <hip/hip_runtime.h>
#include <math.h>

typedef __attribute__((ext_vector_type(16))) _Float16 v16h;
typedef __attribute__((ext_vector_type(8)))  float    v8f;

#define NEXP 16
#define TOPK 4
#define NB   128
#define CIN  32
#define COUT 32
#define HH   64
#define WW   64
#define KDIM 288          // CIN * 9
#define RPB  4            // output rows per block
#define XROWS (RPB + 2)   // staged input rows (halo)
#define CELL 40           // halves per (row,col) cell: 32 ch + 8 pad (80B, bank-skewed)
#define XSSZ (XROWS * 66 * CELL)      // 15840 halves
#define AWSZ (TOPK * COUT * KDIM)     // 36864 halves

// ---------------------------------------------------------------------------
// Kernel 1: gating. One block per batch image.
// ---------------------------------------------------------------------------
__global__ __launch_bounds__(256) void gate_kernel(
    const float* __restrict__ x,
    const float* __restrict__ w_gate,
    float* __restrict__ gates,      // [NB][TOPK]
    int*   __restrict__ idxs) {     // [NB][TOPK]
  __shared__ float part[256];
  __shared__ float gxs[CIN];
  __shared__ float lg[NEXP];
  const int b   = blockIdx.x;
  const int tid = threadIdx.x;
  const int c   = tid & 31;
  const int g   = tid >> 5;
  const float* xp = x + ((size_t)b * CIN + c) * (HH * WW) + g * 512;
  float s = 0.f;
  for (int p = 0; p < 512; ++p) s += xp[p];
  part[tid] = s;
  __syncthreads();
  if (tid < CIN) {
    float t = 0.f;
    for (int gg = 0; gg < 8; ++gg) t += part[tid + 32 * gg];
    gxs[tid] = t * (1.0f / (HH * WW));
  }
  __syncthreads();
  if (tid < NEXP) {
    float l = 0.f;
    for (int cc = 0; cc < CIN; ++cc) l += gxs[cc] * w_gate[cc * NEXP + tid];
    lg[tid] = l;
  }
  __syncthreads();
  if (tid == 0) {
    float v[NEXP];
    for (int e = 0; e < NEXP; ++e) v[e] = lg[e];
    int   sel[TOPK];
    float sv[TOPK];
    for (int k = 0; k < TOPK; ++k) {
      int best = 0; float bv = v[0];
      for (int e = 1; e < NEXP; ++e) if (v[e] > bv) { bv = v[e]; best = e; }
      sel[k] = best; sv[k] = bv; v[best] = -INFINITY;
    }
    float m = sv[0], ssum = 0.f, ex[TOPK];
    for (int k = 0; k < TOPK; ++k) { ex[k] = expf(sv[k] - m); ssum += ex[k]; }
    for (int k = 0; k < TOPK; ++k) {
      gates[b * TOPK + k] = ex[k] / ssum;
      idxs [b * TOPK + k] = sel[k];
    }
  }
}

// ---------------------------------------------------------------------------
// Kernel 2: loss. Single thread, fully deterministic (no atomics).
// ---------------------------------------------------------------------------
__global__ void loss_kernel(const float* __restrict__ gates,
                            const int* __restrict__ idxs,
                            float* __restrict__ out_loss) {
  if (threadIdx.x != 0 || blockIdx.x != 0) return;
  float imp[NEXP], ld[NEXP];
  for (int e = 0; e < NEXP; ++e) { imp[e] = 0.f; ld[e] = 0.f; }
  for (int b = 0; b < NB; ++b)
    for (int k = 0; k < TOPK; ++k) {
      int e = idxs[b * TOPK + k];
      imp[e] += gates[b * TOPK + k];
      ld[e]  += 1.0f;
    }
  float loss = 0.f;
  for (int a = 0; a < 2; ++a) {
    const float* v = a ? ld : imp;
    float mean = 0.f;
    for (int e = 0; e < NEXP; ++e) mean += v[e];
    mean *= (1.0f / NEXP);
    float var = 0.f;
    for (int e = 0; e < NEXP; ++e) { float d = v[e] - mean; var += d * d; }
    var *= (1.0f / (NEXP - 1));   // ddof=1
    loss += var / (mean * mean + 1e-10f);
  }
  *out_loss = loss * 0.01f;
}

// ---------------------------------------------------------------------------
// Kernel 3: MoE conv, implicit-GEMM WMMA, 4 output rows per block.
// Grid: (H/RPB, B). Block: 256 threads = 8 waves.
// Wave w: tileM = w&1 (Cout half), tileN = w>>1 (16-pixel group of the row).
// Contraction index K = tap*32 + c (tap = dy*3+dx fixed per 32-chunk).
// LDS (dynamic): aw[4 experts][32][288] f16 (A side, contiguous frag pairs)
//             ++ xs[6 rows][66 cols][CELL] f16, channel-innermost so each
//                B-fragment half is a contiguous 16B run -> ds_load_b128.
// Single barrier per block; LDS is read-only afterwards.
// ---------------------------------------------------------------------------
__global__ __launch_bounds__(256) void moe_conv_kernel(
    const float* __restrict__ x,
    const float* __restrict__ expert_w,   // [E][COUT][CIN][3][3]
    const float* __restrict__ expert_b,   // [E][COUT]
    const float* __restrict__ gates,      // [NB][TOPK]
    const int*   __restrict__ idxs,       // [NB][TOPK]
    float* __restrict__ out) {            // [NB][COUT][H][W]
  extern __shared__ _Float16 smem[];
  _Float16* aw = smem;                    // AWSZ halves
  _Float16* xs = smem + AWSZ;             // XSSZ halves

  const int h0  = blockIdx.x * RPB;
  const int b   = blockIdx.y;
  const int tid = threadIdx.x;

  // Stage the 4 selected experts' weights: (c,tap) -> (tap,c) transpose, f16.
  // Unrolled per-expert phases keep the expert id compile-time indexed.
#pragma unroll
  for (int k = 0; k < TOPK; ++k) {
    const int ek = idxs[b * TOPK + k];
    const float* wsrc = expert_w + (size_t)ek * COUT * KDIM;
    _Float16* awk = aw + k * (COUT * KDIM);
    for (int j = tid; j < COUT * KDIM; j += 256) {
      int m   = j / KDIM;
      int r   = j - m * KDIM;
      int tap = r >> 5;
      int c   = r & 31;
      awk[j] = (_Float16)wsrc[(m * CIN + c) * 9 + tap];
    }
  }

  // Stage 6 input rows, channel-innermost. Source-major loop keeps global
  // reads coalesced; LDS writes scatter across the 80B-skewed cells.
  for (int i = tid; i < CIN * XROWS * WW; i += 256) {
    int c   = i / (XROWS * WW);
    int rem = i - c * (XROWS * WW);
    int row = rem >> 6;          // /WW
    int gw  = rem & 63;          // %WW
    int gh  = h0 + row - 1;
    float v = 0.f;
    if ((unsigned)gh < HH)
      v = x[(((size_t)b * CIN + c) * HH + gh) * WW + gw];
    xs[(row * 66 + gw + 1) * CELL + c] = (_Float16)v;
  }
  // Zero the left/right padding columns.
  for (int j = tid; j < XROWS * 2 * CIN; j += 256) {
    int row  = j >> 6;           // /(2*CIN)
    int rem  = j & 63;
    int side = rem >> 5;
    int c    = rem & 31;
    xs[(row * 66 + (side ? 65 : 0)) * CELL + c] = (_Float16)0.f;
  }
  __syncthreads();   // only barrier: LDS is read-only from here on

  const int lane  = tid & 31;
  const int wave  = tid >> 5;
  const int tileM = wave & 1;
  const int tileN = wave >> 1;
  const int ln16  = lane & 15;
  const int half  = lane >> 4;
  const int wpix  = tileN * 16 + ln16;    // output pixel column (B/D frag col)
  const int mA    = tileM * 16 + ln16;    // A fragment row (cout)

#pragma unroll 1
  for (int rr = 0; rr < RPB; ++rr) {
    const int h = h0 + rr;

    // Build the 9 B fragments (K=32 each); reused across all 4 experts.
    // B 32x16 f16 frag in this code's contiguous k order (matches A): lane col = lane%16; K = (lane/16)*16 + elem.
    // Channel-innermost xs => each half-fragment is 16 contiguous halves.
    v16h bfrag[9];
#pragma unroll
    for (int kc = 0; kc < 9; ++kc) {
      int dy = kc / 3, dx = kc - dy * 3;
      const _Float16* base =
          &xs[((rr + dy) * 66 + wpix + dx) * CELL + half * 16];
      v16h t;
#pragma unroll
      for (int e = 0; e < 16; ++e) t[e] = base[e];   // contiguous -> b128 x2
      bfrag[kc] = t;
    }

    float comb[8];
#pragma unroll
    for (int r = 0; r < 8; ++r) comb[r] = 0.f;

#pragma unroll 1
    for (int k = 0; k < TOPK; ++k) {
      // Per-iteration loads (uniform gate/idx + 8 contiguous bias floats)
      // instead of pre-hoisted arrays: avoids movrels/m0 dynamic indexing.
      const int   e  = idxs [b * TOPK + k];
      const float gk = gates[b * TOPK + k];
      const float* bptr = expert_b + e * COUT + tileM * 16 + half * 8;
      const _Float16* awk = aw + k * (COUT * KDIM);

      v8f acc = {};
#pragma unroll
      for (int kc = 0; kc < 9; ++kc) {
        // A 16x32 f16 frag in this code's contiguous k order (matches B): row = lane%16; half = lane/16:
        //   elem e -> K = 16*half + e
        const _Float16* ar = &awk[mA * KDIM + kc * 32 + half * 16];
        v16h a;
#pragma unroll
        for (int ee = 0; ee < 8; ++ee)  a[ee] = ar[ee];
#pragma unroll
        for (int ee = 8; ee < 16; ++ee) a[ee] = ar[ee];
        acc = __builtin_amdgcn_wmma_f32_16x16x32_f16(
            false, a, false, bfrag[kc], (short)0, acc, false, false);
      }
      // Fused bias + gate*exp(y). D layout: vgpr r -> m = r + 8*half.
#pragma unroll
      for (int r = 0; r < 8; ++r)
        comb[r] += gk * expf(acc[r] + bptr[r]);
    }

#pragma unroll
    for (int r = 0; r < 8; ++r) {
      int cout = tileM * 16 + half * 8 + r;
      float cv = comb[r];
      if (cv == 0.f) cv = 2.220446049250313e-16f;  // np.finfo(float64).eps
      out[(((size_t)b * COUT + cout) * HH + h) * WW + wpix] = logf(cv);
    }
  }
}

// ---------------------------------------------------------------------------
extern "C" void kernel_launch(void* const* d_in, const int* in_sizes, int n_in,
                              void* d_out, int out_size, void* d_ws, size_t ws_size,
                              hipStream_t stream) {
  const float* x        = (const float*)d_in[0];
  const float* w_gate   = (const float*)d_in[1];
  // d_in[2] = w_noise : unused (train == 0 -> eval path)
  const float* expert_w = (const float*)d_in[3];
  const float* expert_b = (const float*)d_in[4];
  // d_in[5] = train (int scalar, 0): eval path hardcoded

  float* out   = (float*)d_out;
  float* wsf   = (float*)d_ws;
  float* gates = wsf;                 // 512 floats
  int*   idxs  = (int*)(wsf + 512);   // 512 ints

  gate_kernel<<<NB, 256, 0, stream>>>(x, w_gate, gates, idxs);

  dim3 grid(HH / RPB, NB);
  size_t shmem = (size_t)(AWSZ + XSSZ) * sizeof(_Float16);  // 105408 B
  moe_conv_kernel<<<grid, 256, shmem, stream>>>(x, expert_w, expert_b,
                                                gates, idxs, out);

  loss_kernel<<<1, 1, 0, stream>>>(gates, idxs, out + (size_t)NB * COUT * HH * WW);
}
